// Encoder_17076789969378
// MI455X (gfx1250) — matrix operand order repaired and checked by reading, not compiled, not run on hardware
//
#include <hip/hip_runtime.h>

typedef _Float16 v16h __attribute__((ext_vector_type(16)));
typedef _Float16 v8h  __attribute__((ext_vector_type(8)));
typedef float    v8f  __attribute__((ext_vector_type(8)));
typedef float    v4f  __attribute__((ext_vector_type(4)));
typedef float    v2f  __attribute__((ext_vector_type(2)));

#define WAVES_PER_BLOCK 8
#define MAX_BLOCKS 1024

// Padded row strides (f16 elements) for the [pt][feat] staging buffers.
#define S1 40   // h1: 16 feats + K=16 bias row (1.0) + zeros to 32 + pad
#define S2 40   // h2: 32 feats + pad
#define S3 72   // h3: 64 feats + pad

struct WaveBuf {
    _Float16 h1[16 * S1];
    _Float16 h2[16 * S2];
    _Float16 h3[16 * S3];
};  // 4864 B per wave (reused as 128-float reduction slot at the end)

// Unified weight A-fragment store: frags 0-1 = L2, 2-5 = L3, 6-21 = L4.
#define NFRAG 22

__device__ __forceinline__ v8f wmma_f16(v16h a, v16h b, v8f c) {
    return __builtin_amdgcn_wmma_f32_16x16x32_f16(
        false, a, false, b, (short)0, c, false, false);
}

#define CAT16(lo, hi) __builtin_shufflevector(lo, hi, 0,1,2,3,4,5,6,7,8,9,10,11,12,13,14,15)

// B-activation fragment (K=32 x N=16pts) from a [pt][feat] buffer:
// lane: N = pt = lnm, K = kbase + hlf*8 + 0..7, then kbase + 16 + hlf*8 + 0..7.
template <int S>
__device__ __forceinline__ v16h loadB(const _Float16* h, int lnm, int hlf, int kbase) {
    const v8h lo = *(const v8h*)&h[lnm * S + kbase + hlf * 8];
    const v8h hi = *(const v8h*)&h[lnm * S + kbase + 16 + hlf * 8];
    return CAT16(lo, hi);
}

__global__ __launch_bounds__(256, 4)   // >=4 waves/SIMD -> VGPRs capped at 256
void mlp_mean_kernel(const float* __restrict__ pts,
                     const float* __restrict__ W1, const float* __restrict__ B1,
                     const float* __restrict__ W2, const float* __restrict__ B2,
                     const float* __restrict__ W3, const float* __restrict__ B3,
                     const float* __restrict__ W4,
                     float* __restrict__ ws, int n_points)
{
    __shared__ __align__(16) WaveBuf buf[WAVES_PER_BLOCK];          // 38 KB
    __shared__ __align__(16) _Float16 wlo[NFRAG][32][8];            // 11 KB
    __shared__ __align__(16) _Float16 whi[NFRAG][32][8];            // 11 KB

    const int tid  = threadIdx.x;
    const int wid  = tid >> 5;
    const int lane = tid & 31;
    const int hlf  = lane >> 4;   // 0: lanes 0-15, 1: lanes 16-31
    const int lnm  = lane & 15;

    // ---- cooperative one-time fill of all weight A-fragments in LDS ----
    // A-frag (16x32 f16): lane l: M = out-in-tile = l&15;
    // halves j<8: K = hl*8 + j ; j>=8: K = 16 + hl*8 + (j-8).
    for (int p = tid; p < NFRAG * 32; p += 256) {
        int f   = p >> 5;
        int l   = p & 31;
        int hl  = l >> 4;
        int m   = l & 15;
        #pragma unroll 1
        for (int j = 0; j < 8; ++j) {
            int klo = hl * 8 + j;
            int khi = 16 + hl * 8 + j;
            _Float16 vlo, vhi;
            if (f < 2) {                       // layer 2: W2[32][16] + bias col K=16
                int out = f * 16 + m;
                vlo = (_Float16)W2[out * 16 + klo];
                vhi = (khi == 16) ? (_Float16)B2[out] : (_Float16)0.0f;
            } else if (f < 6) {                // layer 3: W3[64][32]
                int out = (f - 2) * 16 + m;
                vlo = (_Float16)W3[out * 32 + klo];
                vhi = (_Float16)W3[out * 32 + khi];
            } else {                           // layer 4: W4[128][64], f-6 = kt*8+nt
                int g   = f - 6;
                int out = (g & 7) * 16 + m;
                int kb  = (g >> 3) * 32;
                vlo = (_Float16)W4[out * 64 + kb + klo];
                vhi = (_Float16)W4[out * 64 + kb + khi];
            }
            wlo[f][l][j] = vlo;
            whi[f][l][j] = vhi;
        }
    }

    // ---- layer-1 fragments (fp32 WMMA 16x16x4): A = weights + bias row ----
    v2f w1A;
    w1A.x = hlf ? W1[lnm * 3 + 2] : W1[lnm * 3 + 0];
    w1A.y = hlf ? B1[lnm]         : W1[lnm * 3 + 1];

    // ---- layer-3 bias C-vectors (feat = nt*16 + r + 8*hlf) ----
    v8f cb3[4];
    #pragma unroll
    for (int nt = 0; nt < 4; ++nt)
        #pragma unroll
        for (int r = 0; r < 8; ++r)
            cb3[nt][r] = B3[nt * 16 + 8 * hlf + r];

    v8f acc4[8] = {};          // persistent layer-4 accumulators (bias at finalize)
    const v8h z8 = {};
    const v8f zc = {};

    _Float16* h1 = buf[wid].h1;
    _Float16* h2 = buf[wid].h2;
    _Float16* h3 = buf[wid].h3;

    // One-time: h1 pad carries the layer-2 bias activation: feat16 = 1.0,
    // feats 17..31 = 0.  Per-tile stores only touch feats 0..15.
    {
        v8h pad = z8;
        if (hlf == 0) pad[0] = (_Float16)1.0f;
        *(v8h*)&h1[lnm * S1 + 16 + hlf * 8] = pad;
    }

    const int n_tiles = n_points >> 4;
    const int stride  = gridDim.x * WAVES_PER_BLOCK;

    __syncthreads();   // weight fragments visible to all waves

    for (int t = blockIdx.x * WAVES_PER_BLOCK + wid; t < n_tiles; t += stride) {
        // Opaque per-iteration redefinition of the fragment offset: stops the
        // compiler from hoisting the (loop-invariant) LDS weight loads into
        // registers and spilling them to scratch. Zero instructions emitted.
        int wofs = lane * 8;
        asm volatile("" : "+v"(wofs));
        const _Float16* wloP = &wlo[0][0][0] + wofs;   // frag f at +f*256 elems
        const _Float16* whiP = &whi[0][0][0] + wofs;

        int tn = t + stride;
        if (tn < n_tiles)
            __builtin_prefetch(pts + (size_t)(tn * 16 + lnm) * 3, 0, 0);

        // ---- Layer 1 (3->16): D = W1aug x [x,y,z,1] = [feat x pt] ----
        const float* pp = pts + (size_t)(t * 16 + lnm) * 3;
        v2f p1;                              // B f32 4x16: v0: K=0/K=2; v1: K=1/K=3
        p1.x = hlf ? pp[2] : pp[0];
        p1.y = hlf ? 1.0f  : pp[1];
        v8f d1 = __builtin_amdgcn_wmma_f32_16x16x4_f32(
            false, w1A, false, p1, (short)0, zc, false, false);
        v8h hv;                              // lane: feats 8*hlf..+7 of pt lnm
        #pragma unroll
        for (int r = 0; r < 8; ++r) hv[r] = (_Float16)d1[r];
        hv = __builtin_elementwise_max(hv, z8);
        *(v8h*)&h1[lnm * S1 + 8 * hlf] = hv;

        // ---- Layer 2 (16->32), K=32 with bias row ----
        v16h bAct2 = loadB<S1>(h1, lnm, hlf, 0);   // hlf=0 hi reads K16..23 incl 1.0
        #pragma unroll
        for (int nt = 0; nt < 2; ++nt) {
            v16h a = CAT16(*(const v8h*)&wloP[nt * 256],
                           *(const v8h*)&whiP[nt * 256]);
            v8f d = wmma_f16(a, bAct2, zc);
            v8h hw;
            #pragma unroll
            for (int r = 0; r < 8; ++r) hw[r] = (_Float16)d[r];
            hw = __builtin_elementwise_max(hw, z8);
            *(v8h*)&h2[lnm * S2 + nt * 16 + 8 * hlf] = hw;
        }

        // ---- Layer 3 (32->64), K=32 exact, bias via C operand ----
        v16h bAct3 = loadB<S2>(h2, lnm, hlf, 0);
        #pragma unroll
        for (int nt = 0; nt < 4; ++nt) {
            v16h a = CAT16(*(const v8h*)&wloP[(2 + nt) * 256],
                           *(const v8h*)&whiP[(2 + nt) * 256]);
            v8f d = wmma_f16(a, bAct3, cb3[nt]);
            v8h hw;
            #pragma unroll
            for (int r = 0; r < 8; ++r) hw[r] = (_Float16)d[r];
            hw = __builtin_elementwise_max(hw, z8);
            *(v8h*)&h3[lnm * S3 + nt * 16 + 8 * hlf] = hw;
        }

        // ---- Layer 4 (64->128), K=64 = 2 K-tiles; accumulate across tiles ----
        v16h bAct4[2];
        #pragma unroll
        for (int kt = 0; kt < 2; ++kt)
            bAct4[kt] = loadB<S3>(h3, lnm, hlf, kt * 32);
        #pragma unroll
        for (int nt = 0; nt < 8; ++nt) {
            #pragma unroll
            for (int kt = 0; kt < 2; ++kt) {
                int f = 6 + kt * 8 + nt;
                v16h a = CAT16(*(const v8h*)&wloP[f * 256],
                               *(const v8h*)&whiP[f * 256]);
                acc4[nt] = wmma_f16(a, bAct4[kt], acc4[nt]);
            }
        }
    }

    // ---------------- reduction (per-wave, fixed order) ----------------
    // acc4[nt][r] on lane (lnm,hlf) = partial for feat nt*16 + r + 8*hlf,
    // summed over the 16 lanes of the half-wave via non-returning ds_add_f32
    // into the wave's own (now idle) staging buffer.
    float* red = (float*)&buf[wid];
    *(v4f*)&red[lane * 4] = (v4f){};          // zero own 128-float slot
    #pragma unroll
    for (int nt = 0; nt < 8; ++nt)
        #pragma unroll
        for (int r = 0; r < 8; ++r)
            atomicAdd(&red[nt * 16 + r + 8 * hlf], acc4[nt][r]);
    __syncthreads();
    if (tid < 128) {
        float s = 0.0f;
        #pragma unroll
        for (int w = 0; w < WAVES_PER_BLOCK; ++w)
            s += ((const float*)&buf[w])[tid];
        ws[(size_t)blockIdx.x * 128 + tid] = s;   // fixed-order per-block partial
    }
}

__global__ void finalize_kernel(const float* __restrict__ ws,
                                const float* __restrict__ B4,
                                float* __restrict__ out,
                                int n_blocks, int n_points)
{
    int f = threadIdx.x;
    if (f < 128) {
        float s = 0.0f;
        for (int b = 0; b < n_blocks; ++b) s += ws[(size_t)b * 128 + f];
        out[f] = s / (float)n_points + B4[f];   // bias applied once post-mean
    }
}

extern "C" void kernel_launch(void* const* d_in, const int* in_sizes, int n_in,
                              void* d_out, int out_size, void* d_ws, size_t ws_size,
                              hipStream_t stream) {
    (void)n_in; (void)out_size;
    const float* pts = (const float*)d_in[0];
    const float* W1  = (const float*)d_in[1];
    const float* B1  = (const float*)d_in[2];
    const float* W2  = (const float*)d_in[3];
    const float* B2  = (const float*)d_in[4];
    const float* W3  = (const float*)d_in[5];
    const float* B3  = (const float*)d_in[6];
    const float* W4  = (const float*)d_in[7];
    const float* B4  = (const float*)d_in[8];

    int n_points = in_sizes[0] / 3;

    size_t max_blocks = ws_size / (128 * sizeof(float));
    int nb = (int)(max_blocks < (size_t)MAX_BLOCKS ? max_blocks : (size_t)MAX_BLOCKS);
    if (nb < 1) nb = 1;

    mlp_mean_kernel<<<nb, 256, 0, stream>>>(pts, W1, B1, W2, B2, W3, B3, W4,
                                            (float*)d_ws, n_points);
    finalize_kernel<<<1, 128, 0, stream>>>((const float*)d_ws, B4,
                                           (float*)d_out, nb, n_points);
}
